// NeighborSelfAttention_39702677684464
// MI455X (gfx1250) — matrix operand order repaired and checked by reading, not compiled, not run on hardware
//
#include <hip/hip_runtime.h>

// ---------------------------------------------------------------------------
// NeighborSelfAttention, fused, f16 WMMA (gfx1250 / CDNA5, wave32)
//
// Exact algebraic reductions used (no approximation beyond f16 operands):
//   scores[q,k] = x_q^T (Wq^T Wk) x_k + (Wk^T bq)·x_k  (+ row-const, dropped)
//   aggregated  = Wv (p^T x) + bv,  p = column-mean of softmax rows (sum p = 1)
// bk only contributes a per-row constant to scores -> irrelevant to softmax.
//
// CDNA5 paths: v_wmma_f32_16x16x32_f16 for all GEMMs; Mt weight matrix staged
// into LDS with GLOBAL_LOAD_ASYNC_TO_LDS_B128 (ASYNCcnt), overlapped with the
// per-wave x staging, completed with s_wait_asynccnt before the barrier.
// ---------------------------------------------------------------------------

typedef _Float16 v8h  __attribute__((ext_vector_type(8)));
typedef _Float16 v16h __attribute__((ext_vector_type(16)));
typedef float    v8f  __attribute__((ext_vector_type(8)));

constexpr int DIM  = 256;   // feature dim
constexpr int KNB  = 16;    // neighbors per point
constexpr int PPB  = 8;     // points per block (= waves per block)
constexpr int NT   = 256;   // threads per block (8 wave32)

// workspace byte offsets
constexpr int WS_MT = 0;        // Mt[e'][d] = (Wq^T Wk)[d][e']  f16, 256x256
constexpr int WS_WV = 131072;   // Wv row-major f16
constexpr int WS_WO = 262144;   // Wo row-major f16
constexpr int WS_U  = 393216;   // u[d'] = (Wk^T bq)[d'] f32

// LDS byte offsets
constexpr int L_MT  = 0;        // 131072
constexpr int L_X   = 131072;   // 8 * 16*256 f16 = 65536
constexpr int L_T   = 196608;   // 65536  (t' per point, f16)
constexpr int L_PV  = 262144;   // 8*16 f32 = 512
constexpr int L_Y   = 262656;   // 16*256 f16 = 8192 (rows 8..15 zero pad)
constexpr int L_AGG = 270848;   // 8192
constexpr int L_O   = 279040;   // 16*256 f32 = 16384
constexpr int L_TOT = 295424;   // < 320KB/WGP

// ---- prep kernels ---------------------------------------------------------

__global__ void prep_mt(const float* __restrict__ Wq, const float* __restrict__ Wk,
                        _Float16* __restrict__ Mt) {
  // Mt[e'][d] = sum_f Wq[f][d] * Wk[f][e']   (row = WMMA B-operand N index)
  const int ep = blockIdx.x;
  const int d  = threadIdx.x;
  float acc = 0.f;
  for (int f = 0; f < DIM; ++f)
    acc += Wq[f * DIM + d] * Wk[f * DIM + ep];
  Mt[ep * DIM + d] = (_Float16)acc;
}

__global__ void prep_misc(const float* __restrict__ Wv, const float* __restrict__ Wo,
                          const float* __restrict__ Wk, const float* __restrict__ bq,
                          _Float16* __restrict__ WvH, _Float16* __restrict__ WoH,
                          float* __restrict__ u) {
  const int gid = blockIdx.x * blockDim.x + threadIdx.x;
  if (gid < DIM * DIM) {
    WvH[gid] = (_Float16)Wv[gid];
  } else if (gid < 2 * DIM * DIM) {
    WoH[gid - DIM * DIM] = (_Float16)Wo[gid - DIM * DIM];
  } else if (gid < 2 * DIM * DIM + DIM) {
    const int e = gid - 2 * DIM * DIM;
    float acc = 0.f;
    for (int f = 0; f < DIM; ++f) acc += bq[f] * Wk[f * DIM + e];
    u[e] = acc;
  }
}

// ---- WMMA operand tile loads (row-major source, ld = 256) -----------------
// A 16x32 f16: lanes 0-15 row=lane, K d0..d0+15 ; lanes 16-31: K d0+16..31
__device__ __forceinline__ v16h loadA(const _Float16* base, int lane, int d0) {
  const int row = lane & 15;
  const int dd  = d0 + ((lane >> 4) << 4);
  const _Float16* p = base + row * DIM + dd;
  union { v16h v; v8h h[2]; } uu;
  uu.h[0] = *(const v8h*)(p);
  uu.h[1] = *(const v8h*)(p + 8);
  return uu.v;
}
// B 32x16 f16: lane N = e0+(lane&15); lanes 0-15 hold K d0..d0+15, lanes 16-31 K d0+16..31
__device__ __forceinline__ v16h loadB(const _Float16* base, int lane, int e0, int d0) {
  const int row = e0 + (lane & 15);
  const int dd  = d0 + ((lane >> 4) << 4);
  const _Float16* p = base + row * DIM + dd;
  union { v16h v; v8h h[2]; } uu;
  uu.h[0] = *(const v8h*)(p);
  uu.h[1] = *(const v8h*)(p + 8);
  return uu.v;
}

#define WMMA_F16(a, b, c) \
  __builtin_amdgcn_wmma_f32_16x16x32_f16(false, (a), false, (b), (short)0, (c), false, false)

// ---- fused kernel ---------------------------------------------------------

__global__ void __launch_bounds__(NT)
fused_attn(const float* __restrict__ x,
           const float* __restrict__ bv, const float* __restrict__ bo,
           const float* __restrict__ gamma, const float* __restrict__ beta,
           const _Float16* __restrict__ Mt, const _Float16* __restrict__ WvH,
           const _Float16* __restrict__ WoH, const float* __restrict__ u,
           float* __restrict__ out) {
  extern __shared__ char smem[];
  _Float16* sMT  = (_Float16*)(smem + L_MT);
  _Float16* sX   = (_Float16*)(smem + L_X);
  _Float16* sT   = (_Float16*)(smem + L_T);
  float*    sPV  = (float*)(smem + L_PV);
  _Float16* sY   = (_Float16*)(smem + L_Y);
  _Float16* sAGG = (_Float16*)(smem + L_AGG);
  float*    sO   = (float*)(smem + L_O);

  const int tid = threadIdx.x;
  const int w   = tid >> 5;        // wave id == local point id
  const int L   = tid & 31;
  const int ln  = L & 15;
  const int hi8 = (L >> 4) << 3;   // C/D: M = r + hi8, N = n0 + ln
  const long g  = (long)blockIdx.x * PPB + w;   // global point

  // Phase 0: async-DMA Mt into LDS (256 thr * 32 * 16B = 131072B, ASYNCcnt),
  // overlapped with Phase 1; also zero Y pad rows 8..15.
  {
    // LDS byte address = low 32 bits of generic pointer (ISA: addr[31:0])
    const unsigned ldsb = (unsigned)(uintptr_t)sMT + (unsigned)tid * 16u;
    const unsigned voff = (unsigned)tid * 16u;
#pragma unroll
    for (int i = 0; i < 32; ++i) {
      asm volatile("global_load_async_to_lds_b128 %0, %1, %2"
                   :: "v"(ldsb + (unsigned)i * 4096u),
                      "v"(voff + (unsigned)i * 4096u),
                      "s"(Mt)
                   : "memory");
    }
    v8h z = {};
    ((v8h*)sY)[256 + tid] = z;
  }

  // Phase 1: stage own point's x (f32 -> f16) + residual mean (kept in regs)
  float res[8];
#pragma unroll
  for (int j = 0; j < 8; ++j) res[j] = 0.f;
  {
    const float* xg = x + g * (KNB * DIM);
    for (int k = 0; k < KNB; ++k) {
      const float4 a = *(const float4*)(xg + k * DIM + L * 8);
      const float4 b = *(const float4*)(xg + k * DIM + L * 8 + 4);
      float v[8] = {a.x, a.y, a.z, a.w, b.x, b.y, b.z, b.w};
      v8h h;
#pragma unroll
      for (int j = 0; j < 8; ++j) { res[j] += v[j]; h[j] = (_Float16)v[j]; }
      *(v8h*)(sX + w * (KNB * DIM) + k * DIM + L * 8) = h;
    }
#pragma unroll
    for (int j = 0; j < 8; ++j) res[j] *= (1.f / 16.f);
  }
  // our async Mt writes must land before we signal the barrier
  asm volatile("s_wait_asynccnt 0x0" ::: "memory");
  __syncthreads();

  // Phase 2: t' = x @ M + u, cooperative: wave owns N-strip et={2w,2w+1} for
  // all 8 points; register-block B over 4 M-tiles x 2 N-tiles (4x less LDS B).
  for (int half = 0; half < 2; ++half) {
    v8f acc[4][2] = {};
#pragma unroll
    for (int kt = 0; kt < 8; ++kt) {
      const v16h b0 = loadB(sMT, L, (2 * w + 0) * 16, kt * 32);
      const v16h b1 = loadB(sMT, L, (2 * w + 1) * 16, kt * 32);
#pragma unroll
      for (int m = 0; m < 4; ++m) {
        const v16h a = loadA(sX + (half * 4 + m) * (KNB * DIM), L, kt * 32);
        acc[m][0] = WMMA_F16(a, b0, acc[m][0]);
        acc[m][1] = WMMA_F16(a, b1, acc[m][1]);
      }
    }
#pragma unroll
    for (int m = 0; m < 4; ++m) {
      const int pt = half * 4 + m;
#pragma unroll
      for (int e = 0; e < 2; ++e) {
        const int n = (2 * w + e) * 16 + ln;
        const float uv = u[n];
#pragma unroll
        for (int r = 0; r < 8; ++r)
          sT[pt * (KNB * DIM) + (r + hi8) * DIM + n] = (_Float16)(acc[m][e][r] + uv);
      }
    }
  }
  __syncthreads();

  // Phase 3: scores = (1/16) * t' @ x^T  (16x16, own point)
  const _Float16* myX = sX + w * (KNB * DIM);
  v8f s = {};
  {
    const _Float16* myT = sT + w * (KNB * DIM);
#pragma unroll
    for (int kt = 0; kt < 8; ++kt) {
      const v16h a = loadA(myT, L, kt * 32);
      const v16h b = loadB(myX, L, 0, kt * 32);   // B rows = neighbors
      s = WMMA_F16(a, b, s);
    }
#pragma unroll
    for (int r = 0; r < 8; ++r) s[r] *= 0.0625f;
  }

  // Phase 4: row softmax (16-lane half reductions) -> p = column mean
  float pacc = 0.f;
#pragma unroll
  for (int r = 0; r < 8; ++r) {
    float m = s[r];
    m = fmaxf(m, __shfl_xor(m, 1, 32));
    m = fmaxf(m, __shfl_xor(m, 2, 32));
    m = fmaxf(m, __shfl_xor(m, 4, 32));
    m = fmaxf(m, __shfl_xor(m, 8, 32));
    const float e = __expf(s[r] - m);
    float sm = e;
    sm += __shfl_xor(sm, 1, 32);
    sm += __shfl_xor(sm, 2, 32);
    sm += __shfl_xor(sm, 4, 32);
    sm += __shfl_xor(sm, 8, 32);
    pacc += e / sm;
  }
  pacc += __shfl_xor(pacc, 16, 32);
  pacc *= (1.f / 16.f);
  if (L < 16) sPV[w * 16 + L] = pacc;

  // Phase 5: y = p^T x  (256 vec), f16 into Y row w
  {
    float y[8];
#pragma unroll
    for (int j = 0; j < 8; ++j) y[j] = 0.f;
    for (int k = 0; k < KNB; ++k) {
      const float pk = sPV[w * 16 + k];
      const v8h xv = *(const v8h*)(myX + k * DIM + L * 8);
#pragma unroll
      for (int j = 0; j < 8; ++j) y[j] += pk * (float)xv[j];
    }
    v8h h;
#pragma unroll
    for (int j = 0; j < 8; ++j) h[j] = (_Float16)y[j];
    *(v8h*)(sY + w * DIM + L * 8) = h;
  }
  __syncthreads();

  // Phase 6: AGG = Y @ Wv^T + bv   (waves split the 16 col-tiles)
  for (int et = 2 * w; et < 2 * w + 2; ++et) {
    v8f c = {};
#pragma unroll
    for (int kt = 0; kt < 8; ++kt) {
      const v16h a = loadA(sY, L, kt * 32);
      const v16h b = loadB(WvH, L, et * 16, kt * 32);
      c = WMMA_F16(a, b, c);
    }
    const int n = et * 16 + ln;
    const float bvv = bv[n];
#pragma unroll
    for (int r = 0; r < 8; ++r)
      sAGG[(r + hi8) * DIM + n] = (_Float16)(c[r] + bvv);
  }
  __syncthreads();

  // Phase 7: O = AGG @ Wo^T + bo   (f32 into sO)
  for (int et = 2 * w; et < 2 * w + 2; ++et) {
    v8f c = {};
#pragma unroll
    for (int kt = 0; kt < 8; ++kt) {
      const v16h a = loadA(sAGG, L, kt * 32);
      const v16h b = loadB(WoH, L, et * 16, kt * 32);
      c = WMMA_F16(a, b, c);
    }
    const int n = et * 16 + ln;
    const float bov = bo[n];
#pragma unroll
    for (int r = 0; r < 8; ++r)
      sO[(r + hi8) * DIM + n] = c[r] + bov;
  }
  __syncthreads();

  // Phase 8: + residual, LayerNorm over 256, store
  {
    float o[8];
    float s1 = 0.f, s2 = 0.f;
#pragma unroll
    for (int j = 0; j < 8; ++j) {
      o[j] = sO[w * DIM + L * 8 + j] + res[j];
      s1 += o[j];
      s2 += o[j] * o[j];
    }
#pragma unroll
    for (int msk = 1; msk <= 16; msk <<= 1) {
      s1 += __shfl_xor(s1, msk, 32);
      s2 += __shfl_xor(s2, msk, 32);
    }
    const float mu  = s1 * (1.f / 256.f);
    const float var = s2 * (1.f / 256.f) - mu * mu;
    const float rs  = rsqrtf(var + 1e-5f);
    float t[8];
#pragma unroll
    for (int j = 0; j < 8; ++j)
      t[j] = (o[j] - mu) * rs * gamma[L * 8 + j] + beta[L * 8 + j];
    float* og = out + g * DIM + L * 8;
    *(float4*)(og)     = make_float4(t[0], t[1], t[2], t[3]);
    *(float4*)(og + 4) = make_float4(t[4], t[5], t[6], t[7]);
  }
}

// ---- host glue ------------------------------------------------------------

extern "C" void kernel_launch(void* const* d_in, const int* in_sizes, int n_in,
                              void* d_out, int out_size, void* d_ws, size_t ws_size,
                              hipStream_t stream) {
  const float* x     = (const float*)d_in[0];
  const float* Wq    = (const float*)d_in[1];
  const float* bq    = (const float*)d_in[2];
  const float* Wk    = (const float*)d_in[3];
  // d_in[4] = bk: only shifts every score in a softmax row -> provably unused
  const float* Wv    = (const float*)d_in[5];
  const float* bv    = (const float*)d_in[6];
  const float* Wo    = (const float*)d_in[7];
  const float* bo    = (const float*)d_in[8];
  const float* gamma = (const float*)d_in[9];
  const float* beta  = (const float*)d_in[10];
  float* out = (float*)d_out;

  char* ws = (char*)d_ws;
  _Float16* Mt  = (_Float16*)(ws + WS_MT);
  _Float16* WvH = (_Float16*)(ws + WS_WV);
  _Float16* WoH = (_Float16*)(ws + WS_WO);
  float*    u   = (float*)(ws + WS_U);

  const int N = in_sizes[0] / (KNB * DIM);

  prep_mt<<<DIM, DIM, 0, stream>>>(Wq, Wk, Mt);
  prep_misc<<<(2 * DIM * DIM + DIM + 255) / 256, 256, 0, stream>>>(Wv, Wo, Wk, bq, WvH, WoH, u);
  fused_attn<<<N / PPB, NT, L_TOT, stream>>>(x, bv, bo, gamma, beta, Mt, WvH, WoH, u, out);

  (void)n_in; (void)out_size; (void)ws_size;
}
